// MoERouter_72816875536956
// MI455X (gfx1250) — matrix operand order repaired and checked by reading, not compiled, not run on hardware
//
#include <hip/hip_runtime.h>

// MoE router for MI455X (gfx1250): B=4, S=4096, H=4096, E=64, K=2
// Memory-bound (256MB hidden read @ 23.3TB/s ~= 11.4us floor).
// GEMM via v_wmma_f32_16x16x32_bf16 (fp32 accumulate), LDS-staged tiles,
// fragment loads software-pipelined one sub-chunk ahead.

typedef __attribute__((ext_vector_type(16))) __bf16 v16bf;
typedef __attribute__((ext_vector_type(8)))  __bf16 bf16x8;
typedef __attribute__((ext_vector_type(8)))  float  v8f;

#define N_TOK        16384
#define HID          4096
#define NE           64
#define TOPK         2
#define KC           256                 // K chunk staged in LDS per iteration
#define NSUB         (KC / 32)           // 8 WMMA sub-chunks per K chunk
#define TOK_PER_BLK  64
#define NBLK         (N_TOK / TOK_PER_BLK)   // 256
#define LDA          (KC + 8)            // padded bf16 row stride (264) -> conflict-free frag reads

#define DM_ELEMS     (N_TOK * TOPK * NE)         // 2,097,152 dispatch-mask floats
#define CW_OFF       DM_ELEMS                    // combine_weights offset
#define AUX_OFF      (DM_ELEMS + N_TOK * TOPK)   // 2,129,920 aux scalar offset

__global__ __launch_bounds__(128)
void moe_router_kernel(const float* __restrict__ hidden,
                       const float* __restrict__ W,
                       float* __restrict__ out,
                       float* __restrict__ partial)
{
    // A tile rows 0..63, B tile rows 64..127: 128 * 264 * 2B = 67,584 B.
    // The logits buffer (64 x 65 floats = 16,640 B) is aliased on top after the
    // GEMM (barrier-separated) -> ~67.8 KB total, 4 blocks per WGP.
    __shared__ __align__(16) __bf16 lAB[TOK_PER_BLK + NE][LDA];
    __shared__ float sExp[NE];

    float (*lLog)[NE + 1] = (float (*)[NE + 1])&lAB[0][0];

    const int tid     = threadIdx.x;
    const int lane    = tid & 31;
    const int wave    = tid >> 5;
    const int tokBase = blockIdx.x * TOK_PER_BLK;

    v8f acc[4];
    #pragma unroll
    for (int t = 0; t < 4; ++t)
        #pragma unroll
        for (int j = 0; j < 8; ++j)
            acc[t][j] = 0.0f;

    // WMMA fragment addressing per CDNA5 ISA layouts (wave32):
    //  A 16x32 bf16: lanes 0-15 -> M=lane, K {0..7}U{16..23}; lanes 16-31 -> K {8..15}U{24..31}
    //  B 32x16 bf16: lanes 0-15 -> N=lane, K {0..7}U{16..23}; lanes 16-31 -> N=lane-16, K {8..15}U{24..31}
    const int arow  = wave * 16 + (lane & 15);
    const int akoff = (lane >> 4) << 3;    // 0 or 8
    const int bkoff = (lane >> 4) << 3;    // 0 or 8
    const int bcol  = lane & 15;

    auto loadA = [&](int kb) -> v16bf {
        const bf16x8 x = *(const bf16x8*)&lAB[arow][kb + akoff];
        const bf16x8 y = *(const bf16x8*)&lAB[arow][kb + akoff + 16];
        return __builtin_shufflevector(x, y,
            0, 1, 2, 3, 4, 5, 6, 7, 8, 9, 10, 11, 12, 13, 14, 15);
    };
    auto loadB = [&](int t, int kb) -> v16bf {
        const bf16x8 x = *(const bf16x8*)&lAB[TOK_PER_BLK + t * 16 + bcol][kb + bkoff];
        const bf16x8 y = *(const bf16x8*)&lAB[TOK_PER_BLK + t * 16 + bcol][kb + bkoff + 16];
        return __builtin_shufflevector(x, y,
            0, 1, 2, 3, 4, 5, 6, 7, 8, 9, 10, 11, 12, 13, 14, 15);
    };

    // ---- GEMM: logits[64 tok][64 exp] = hidden * W^T, K = 4096 ----
    for (int k0 = 0; k0 < HID; k0 += KC) {
        __syncthreads();   // protect LDS tiles from previous iteration's readers

        // Stage A (hidden tokens) and B (router weights) as bf16 into LDS.
        // 64 rows x 256 K each = 4096 float4 per matrix; 128 threads x 32 iters.
        #pragma unroll 4
        for (int i = 0; i < (TOK_PER_BLK * KC / 4) / 128; ++i) {
            const int idx = tid + i * 128;
            const int row = idx >> 6;           // 64 float4 per row
            const int c4  = (idx & 63) << 2;    // column in floats

            const float4 va = *(const float4*)&hidden[(size_t)(tokBase + row) * HID + k0 + c4];
            __bf16* da = &lAB[row][c4];
            da[0] = (__bf16)va.x; da[1] = (__bf16)va.y;
            da[2] = (__bf16)va.z; da[3] = (__bf16)va.w;

            const float4 vb = *(const float4*)&W[(size_t)row * HID + k0 + c4];
            __bf16* db = &lAB[TOK_PER_BLK + row][c4];
            db[0] = (__bf16)vb.x; db[1] = (__bf16)vb.y;
            db[2] = (__bf16)vb.z; db[3] = (__bf16)vb.w;

            if (k0 + KC < HID)  // global_prefetch_b8 for next hidden chunk
                __builtin_prefetch(&hidden[(size_t)(tokBase + row) * HID + k0 + KC + c4], 0, 0);
        }
        __syncthreads();

        // Software-pipelined fragment consumption: loads for sub+1 are issued
        // before the 4 WMMAs of sub, so s_wait_dscnt overlaps matrix issue.
        v16bf aC  = loadA(0);
        v16bf bC0 = loadB(0, 0), bC1 = loadB(1, 0), bC2 = loadB(2, 0), bC3 = loadB(3, 0);

        #pragma unroll
        for (int sub = 0; sub < NSUB; ++sub) {
            v16bf aN = aC, bN0 = bC0, bN1 = bC1, bN2 = bC2, bN3 = bC3;
            if (sub + 1 < NSUB) {
                const int kn = (sub + 1) * 32;
                aN  = loadA(kn);
                bN0 = loadB(0, kn); bN1 = loadB(1, kn);
                bN2 = loadB(2, kn); bN3 = loadB(3, kn);
            }
            acc[0] = __builtin_amdgcn_wmma_f32_16x16x32_bf16(false, aC, false, bC0, (short)0, acc[0], false, false);
            acc[1] = __builtin_amdgcn_wmma_f32_16x16x32_bf16(false, aC, false, bC1, (short)0, acc[1], false, false);
            acc[2] = __builtin_amdgcn_wmma_f32_16x16x32_bf16(false, aC, false, bC2, (short)0, acc[2], false, false);
            acc[3] = __builtin_amdgcn_wmma_f32_16x16x32_bf16(false, aC, false, bC3, (short)0, acc[3], false, false);
            aC = aN; bC0 = bN0; bC1 = bN1; bC2 = bN2; bC3 = bN3;
        }
    }

    __syncthreads();   // all waves done reading lAB before aliasing it as lLog

    // ---- Dump logits to LDS (C/D layout: VGPR j -> M=j / j+8; lane -> N) ----
    {
        const int mbase = wave * 16 + ((lane >> 4) << 3);
        const int ncol  = lane & 15;
        #pragma unroll
        for (int t = 0; t < 4; ++t)
            #pragma unroll
            for (int j = 0; j < 8; ++j)
                lLog[mbase + j][t * 16 + ncol] = acc[t][j];
    }
    if (tid < NE) sExp[tid] = 0.0f;
    __syncthreads();

    // ---- Per-token softmax + top-2 + outputs (one thread per token) ----
    if (tid < TOK_PER_BLK) {
        const int tok = tokBase + tid;

        float mx = -3.402823e38f;
        #pragma unroll 8
        for (int e = 0; e < NE; ++e) mx = fmaxf(mx, lLog[tid][e]);

        float z = 0.0f;
        #pragma unroll 8
        for (int e = 0; e < NE; ++e) z += __expf(lLog[tid][e] - mx);

        float v0 = -3.402823e38f, v1 = -3.402823e38f;
        int   i0 = 0, i1 = 0;
        for (int e = 0; e < NE; ++e) {
            const float v = lLog[tid][e];
            if (v > v0)      { v1 = v0; i1 = i0; v0 = v; i0 = e; }
            else if (v > v1) { v1 = v;  i1 = e; }
        }

        const float p0  = __expf(v0 - mx) / z;
        const float p1  = __expf(v1 - mx) / z;
        const float inv = 1.0f / (p0 + p1);
        const float w0  = p0 * inv;
        const float w1  = p1 * inv;

        // combine_weights [n, k]
        out[CW_OFF + tok * 2 + 0] = w0;
        out[CW_OFF + tok * 2 + 1] = w1;

        // dispatch_mask one-hot rows [tok][k=0..1][e=0..63] (thread owns both rows)
        float4* dm = (float4*)(out + (size_t)tok * (TOPK * NE));
        #pragma unroll 8
        for (int q = 0; q < 32; ++q) {
            const int base = q * 4;
            const int hit  = (q < 16) ? i0 : (NE + i1);
            float4 zv;
            zv.x = (base + 0 == hit) ? 1.0f : 0.0f;
            zv.y = (base + 1 == hit) ? 1.0f : 0.0f;
            zv.z = (base + 2 == hit) ? 1.0f : 0.0f;
            zv.w = (base + 3 == hit) ? 1.0f : 0.0f;
            dm[q] = zv;
        }

        atomicAdd(&sExp[i0], w0);   // ds_add_f32
        atomicAdd(&sExp[i1], w1);
    }
    __syncthreads();

    if (tid < NE) partial[(size_t)blockIdx.x * NE + tid] = sExp[tid];
}

// aux = (AUX_W / n) * sum_e S_e^2
__global__ __launch_bounds__(64)
void moe_aux_kernel(const float* __restrict__ partial, float* __restrict__ out)
{
    __shared__ float sq[NE];
    const int e = threadIdx.x;
    float s = 0.0f;
    for (int b = 0; b < NBLK; ++b) s += partial[(size_t)b * NE + e];
    sq[e] = s * s;
    __syncthreads();
    if (e == 0) {
        float tot = 0.0f;
        #pragma unroll
        for (int i = 0; i < NE; ++i) tot += sq[i];
        out[AUX_OFF] = tot * (0.01f / (float)N_TOK);
    }
}

extern "C" void kernel_launch(void* const* d_in, const int* in_sizes, int n_in,
                              void* d_out, int out_size, void* d_ws, size_t ws_size,
                              hipStream_t stream)
{
    const float* hidden  = (const float*)d_in[0];   // [4,4096,4096] fp32
    const float* W       = (const float*)d_in[1];   // [64,4096]     fp32
    float*       out     = (float*)d_out;           // [dm | cw | aux] = 2,129,921 floats
    float*       partial = (float*)d_ws;            // NBLK*NE floats = 64 KB scratch

    moe_router_kernel<<<NBLK, 128, 0, stream>>>(hidden, W, out, partial);
    moe_aux_kernel<<<1, 64, 0, stream>>>(partial, out);
}
